// Seq2Seq_53317724012725
// MI455X (gfx1250) — matrix operand order repaired and checked by reading, not compiled, not run on hardware
//
#include <hip/hip_runtime.h>
#include <hip/hip_bf16.h>
#include <stdint.h>

// Problem dims
#define BB 32
#define TT 128
#define EE 512
#define HH 1024
#define LDD 256
#define LL 5000

typedef __attribute__((ext_vector_type(16))) __bf16 v16bf;
typedef __attribute__((ext_vector_type(8)))  float  v8f;

union Frag16 { v16bf v; uint4 q[2]; };

__device__ __forceinline__ v8f wmma_bf16(v16bf a, v16bf b, v8f c) {
  return __builtin_amdgcn_wmma_f32_16x16x32_bf16(false, a, false, b, (short)0, c,
                                                 false, false);
}

__device__ __forceinline__ unsigned short f2bf(float f) {
  unsigned int u = __float_as_uint(f);
  u += 0x7FFFu + ((u >> 16) & 1u);          // round-to-nearest-even
  return (unsigned short)(u >> 16);
}

__device__ __forceinline__ float fsig(float x) { return 1.f / (1.f + __expf(-x)); }
__device__ __forceinline__ float ftanh(float x) {
  float e = __expf(-2.f * fabsf(x));
  float t = (1.f - e) / (1.f + e);
  return (x < 0.f) ? -t : t;
}

// ---------- prologue kernels ----------

// out[r][0..ka-1] = bf16(A[r]); out[r][ka..ka+kb-1] = bf16(Bm[r])
__global__ void k_cvt_concat(const float* __restrict__ A, int lda, int ka,
                             const float* __restrict__ Bm, int ldb, int kb,
                             unsigned short* __restrict__ out, int rows) {
  int ktot = ka + kb;
  size_t n = (size_t)rows * ktot;
  for (size_t i = (size_t)blockIdx.x * blockDim.x + threadIdx.x; i < n;
       i += (size_t)gridDim.x * blockDim.x) {
    int r = (int)(i / ktot), k = (int)(i % ktot);
    float v = (k < ka) ? A[(size_t)r * lda + k] : Bm[(size_t)r * ldb + (k - ka)];
    out[i] = f2bf(v);
  }
}

__global__ void k_bias_sum(const float* __restrict__ a, const float* __restrict__ b,
                           float* __restrict__ o, int n) {
  int i = blockIdx.x * blockDim.x + threadIdx.x;
  if (i < n) o[i] = a[i] + b[i];
}

// enc_in row r = (t*B+b); token = sentence.flat[r]  (reshape semantics)
__global__ void k_gather_enc(const int* __restrict__ sent,
                             const float* __restrict__ emb,
                             unsigned short* __restrict__ out) {
  size_t n = (size_t)TT * BB * EE;
  for (size_t i = (size_t)blockIdx.x * blockDim.x + threadIdx.x; i < n;
       i += (size_t)gridDim.x * blockDim.x) {
    int r = (int)(i / EE), k = (int)(i % EE);
    int tok = sent[r];
    out[i] = f2bf(emb[(size_t)tok * EE + k]);
  }
}

// dec_in row r = t*B+b: t==0 -> label_emb[1]; else label_emb[label_seq[b][t-1]]
__global__ void k_gather_dec(const int* __restrict__ labels,
                             const float* __restrict__ emb,
                             unsigned short* __restrict__ out) {
  size_t n = (size_t)TT * BB * LDD;
  for (size_t i = (size_t)blockIdx.x * blockDim.x + threadIdx.x; i < n;
       i += (size_t)gridDim.x * blockDim.x) {
    int r = (int)(i / LDD), k = (int)(i % LDD);
    int t = r / BB, b = r % BB;
    int lab = (t == 0) ? 1 : labels[b * TT + (t - 1)];
    out[i] = f2bf(emb[(size_t)lab * LDD + k]);
  }
}

__global__ void k_init_state(unsigned short* __restrict__ h0,
                             float* __restrict__ c) {
  int i = blockIdx.x * blockDim.x + threadIdx.x;
  if (i < BB * HH) { h0[i] = 0; c[i] = 0.f; }
}

// ---------- LSTM step ----------
// gates = [x_t | h] @ [W_ih ; W_hh]^T + b, then activations.
// A tile (32 x ktot bf16) staged once per block in LDS, shared by 8 waves.
// Each wave owns 16 hidden units, computing i/f/g/o for them (8 accumulators).
// B fragments use a 2-slot ping-pong: gate g+1's loads overlap gate g's WMMAs,
// keeping only 16 B VGPRs live (round 2's 4-deep pipeline caused spills).
__global__ void __launch_bounds__(256, 1)
k_lstm_step(const unsigned short* __restrict__ X, int ldx, int kx,
            const unsigned short* __restrict__ Hprev,
            const unsigned short* __restrict__ W, int ktot,
            const float* __restrict__ bias,
            float* __restrict__ Cst,
            unsigned short* __restrict__ Hout) {
  extern __shared__ unsigned short Alds[];   // 32 * ktot bf16

  // cooperative stage of A = [x_t | h] into LDS (16B chunks)
  int nchunks = (32 * ktot) >> 3;
  for (int ci = threadIdx.x; ci < nchunks; ci += blockDim.x) {
    int e = ci << 3;
    int m = e / ktot, k = e % ktot;
    const unsigned short* src = (k < kx) ? (X + (size_t)m * ldx + k)
                                         : (Hprev + (size_t)m * HH + (k - kx));
    *(uint4*)(&Alds[e]) = *(const uint4*)src;
  }
  __syncthreads();

  int wv = blockIdx.x * (blockDim.x >> 5) + (threadIdx.x >> 5);
  int j0 = wv * 16;                           // grid sized so j0 < HH always
  int lane = threadIdx.x & 31;
  int hh = lane >> 4;
  int mr = lane & 15;

  // per-gate weight row base for this lane (B = W^T: column n = W row n)
  const unsigned short* pbBase[4];
#pragma unroll
  for (int g = 0; g < 4; ++g)
    pbBase[g] = W + (size_t)(g * HH + j0 + mr) * ktot + hh * 8;

  v8f acc[4][2];
#pragma unroll
  for (int g = 0; g < 4; ++g)
#pragma unroll
    for (int m = 0; m < 2; ++m)
#pragma unroll
      for (int r = 0; r < 8; ++r) acc[g][m][r] = 0.f;

  // prime: gate 0, k0 = 0
  Frag16 bp0, bp1;
  bp0.q[0] = *(const uint4*)(pbBase[0]);
  bp0.q[1] = *(const uint4*)(pbBase[0] + 16);

#pragma unroll 1
  for (int k0 = 0; k0 < ktot; k0 += 32) {
    // A fragments from LDS (ds_load_b128)
    const unsigned short* pa = Alds + mr * ktot + k0 + hh * 8;
    Frag16 a0, a1;
    a0.q[0] = *(const uint4*)pa;
    a0.q[1] = *(const uint4*)(pa + 16);
    const unsigned short* pa2 = pa + 16 * ktot;
    a1.q[0] = *(const uint4*)pa2;
    a1.q[1] = *(const uint4*)(pa2 + 16);

    int knext = (k0 + 32 < ktot) ? (k0 + 32) : 0;  // last iter: harmless reload
#pragma unroll
    for (int g = 0; g < 4; ++g) {
      int gn = (g + 1) & 3;
      int kk = (g == 3) ? knext : k0;
      Frag16& cur = (g & 1) ? bp1 : bp0;
      Frag16& nxt = (g & 1) ? bp0 : bp1;
      nxt.q[0] = *(const uint4*)(pbBase[gn] + kk);
      nxt.q[1] = *(const uint4*)(pbBase[gn] + kk + 16);
      __builtin_prefetch(pbBase[g] + k0 + 64, 0, 3);
      acc[g][0] = wmma_bf16(a0.v, cur.v, acc[g][0]);
      acc[g][1] = wmma_bf16(a1.v, cur.v, acc[g][1]);
    }
  }

  // epilogue: activations, c update, h (bf16 for next step's A operand)
  int col = j0 + mr;
  float bi = bias[0 * HH + col];
  float bf = bias[1 * HH + col];
  float bg = bias[2 * HH + col];
  float bo = bias[3 * HH + col];
#pragma unroll
  for (int mt = 0; mt < 2; ++mt) {
#pragma unroll
    for (int r = 0; r < 8; ++r) {
      int m = mt * 16 + hh * 8 + r;            // C-tile layout (ISA 7.12.2)
      float gi = acc[0][mt][r] + bi;
      float gf = acc[1][mt][r] + bf;
      float gg = acc[2][mt][r] + bg;
      float go = acc[3][mt][r] + bo;
      size_t idx = (size_t)m * HH + col;
      float cn = fsig(gf) * Cst[idx] + fsig(gi) * ftanh(gg);
      Cst[idx] = cn;
      Hout[idx] = f2bf(fsig(go) * ftanh(cn));
    }
  }
}

// ---------- score GEMM: (4096 x 1024) bf16 @ W_score^T (-> 5000 cols) + b ----
// Each block owns one 32-row A strip (64 KB in LDS) shared by 8 waves; each
// wave produces one 16-wide N tile. B fragments software-pipelined (2 slots).
#define SC_NTILES 313                          // ceil(5000/16)
#define SC_NB 40                               // ceil(313/8)
__global__ void __launch_bounds__(256, 1)
k_score(const unsigned short* __restrict__ A,
        const unsigned short* __restrict__ Wn,
        const float* __restrict__ bias,
        float* __restrict__ out) {
  extern __shared__ unsigned short Alds[];     // 32 x HH bf16 = 64 KB
  int mi = blockIdx.x / SC_NB;
  int nb = blockIdx.x % SC_NB;
  int row0 = mi * 32;

  for (int ci = threadIdx.x; ci < (32 * HH) >> 3; ci += blockDim.x) {
    int e = ci << 3;
    int m = e >> 10, k = e & (HH - 1);
    *(uint4*)(&Alds[e]) = *(const uint4*)(A + (size_t)(row0 + m) * HH + k);
  }
  __syncthreads();

  int ni = nb * 8 + (threadIdx.x >> 5);
  if (ni >= SC_NTILES) return;                 // wave-uniform exit (after barrier)
  int n0 = ni * 16;
  int lane = threadIdx.x & 31;
  int hh = lane >> 4;
  int mr = lane & 15;
  int ncol = n0 + mr;
  bool nok = ncol < LL;

  const unsigned short* pbB = Wn + (size_t)(nok ? ncol : (LL - 1)) * HH + hh * 8;

  v8f acc0, acc1;
#pragma unroll
  for (int r = 0; r < 8; ++r) { acc0[r] = 0.f; acc1[r] = 0.f; }

  uint4 z; z.x = 0; z.y = 0; z.z = 0; z.w = 0;
  Frag16 bcur;
  bcur.q[0] = z; bcur.q[1] = z;
  if (nok) {
    bcur.q[0] = *(const uint4*)pbB;
    bcur.q[1] = *(const uint4*)(pbB + 16);
  }

#pragma unroll 1
  for (int k0 = 0; k0 < HH; k0 += 32) {
    int kn = (k0 + 32 < HH) ? (k0 + 32) : 0;
    Frag16 bnext;
    bnext.q[0] = z; bnext.q[1] = z;
    if (nok) {
      bnext.q[0] = *(const uint4*)(pbB + kn);
      bnext.q[1] = *(const uint4*)(pbB + kn + 16);
      __builtin_prefetch(pbB + k0 + 64, 0, 3);
    }

    const unsigned short* pa = Alds + mr * HH + k0 + hh * 8;
    Frag16 a0, a1;
    a0.q[0] = *(const uint4*)pa;
    a0.q[1] = *(const uint4*)(pa + 16);
    const unsigned short* pa2 = pa + 16 * HH;
    a1.q[0] = *(const uint4*)pa2;
    a1.q[1] = *(const uint4*)(pa2 + 16);

    acc0 = wmma_bf16(a0.v, bcur.v, acc0);
    acc1 = wmma_bf16(a1.v, bcur.v, acc1);
    bcur = bnext;
  }

  if (nok) {
    float bs = bias[ncol];
#pragma unroll
    for (int r = 0; r < 8; ++r) {
      int m = row0 + hh * 8 + r;
      out[(size_t)m * LL + ncol] = acc0[r] + bs;
      out[(size_t)(m + 16) * LL + ncol] = acc1[r] + bs;
    }
  }
}

// ---------- host ----------
extern "C" void kernel_launch(void* const* d_in, const int* in_sizes, int n_in,
                              void* d_out, int out_size, void* d_ws, size_t ws_size,
                              hipStream_t stream) {
  (void)in_sizes; (void)n_in; (void)out_size; (void)ws_size;
  const int*   sentence  = (const int*)d_in[0];
  const int*   label_seq = (const int*)d_in[1];
  const float* word_emb  = (const float*)d_in[2];
  const float* label_emb = (const float*)d_in[3];
  const float* enc_W_ih  = (const float*)d_in[4];
  const float* enc_W_hh  = (const float*)d_in[5];
  const float* enc_b_ih  = (const float*)d_in[6];
  const float* enc_b_hh  = (const float*)d_in[7];
  const float* dec_W_ih  = (const float*)d_in[8];
  const float* dec_W_hh  = (const float*)d_in[9];
  const float* dec_b_ih  = (const float*)d_in[10];
  const float* dec_b_hh  = (const float*)d_in[11];
  const float* W_score   = (const float*)d_in[12];
  const float* b_score   = (const float*)d_in[13];
  float* out = (float*)d_out;

  uint8_t* ws = (uint8_t*)d_ws;
  size_t off = 0;
  auto take = [&](size_t bytes) {
    uint8_t* p = ws + off;
    off += (bytes + 255) & ~(size_t)255;
    return p;
  };
  unsigned short* Xenc  = (unsigned short*)take((size_t)TT * BB * EE * 2);
  unsigned short* Xdec  = (unsigned short*)take((size_t)TT * BB * LDD * 2);
  unsigned short* Wenc  = (unsigned short*)take((size_t)4 * HH * (EE + HH) * 2);
  unsigned short* Wdec  = (unsigned short*)take((size_t)4 * HH * (LDD + HH) * 2);
  unsigned short* Wsc   = (unsigned short*)take((size_t)LL * HH * 2);
  float*          benc  = (float*)take((size_t)4 * HH * 4);
  float*          bdec  = (float*)take((size_t)4 * HH * 4);
  unsigned short* h0buf = (unsigned short*)take((size_t)BB * HH * 2);
  unsigned short* h1buf = (unsigned short*)take((size_t)BB * HH * 2);
  float*          Cst   = (float*)take((size_t)BB * HH * 4);
  unsigned short* decHs = (unsigned short*)take((size_t)TT * BB * HH * 2);

  // prologue: weight conversion/concat, bias sums, embedding gathers, state init
  k_cvt_concat<<<4096, 256, 0, stream>>>(enc_W_ih, EE, EE, enc_W_hh, HH, HH,
                                         Wenc, 4 * HH);
  k_cvt_concat<<<4096, 256, 0, stream>>>(dec_W_ih, LDD, LDD, dec_W_hh, HH, HH,
                                         Wdec, 4 * HH);
  k_cvt_concat<<<4096, 256, 0, stream>>>(W_score, HH, HH, nullptr, 0, 0,
                                         Wsc, LL);
  k_bias_sum<<<16, 256, 0, stream>>>(enc_b_ih, enc_b_hh, benc, 4 * HH);
  k_bias_sum<<<16, 256, 0, stream>>>(dec_b_ih, dec_b_hh, bdec, 4 * HH);
  k_gather_enc<<<4096, 256, 0, stream>>>(sentence, word_emb, Xenc);
  k_gather_dec<<<2048, 256, 0, stream>>>(label_seq, label_emb, Xdec);
  k_init_state<<<(BB * HH + 255) / 256, 256, 0, stream>>>(h0buf, Cst);

  unsigned short* hp[2] = {h0buf, h1buf};
  const size_t ldsEnc = (size_t)32 * (EE + HH) * 2;   // 96 KB
  const size_t ldsDec = (size_t)32 * (LDD + HH) * 2;  // 80 KB

  // encoder: 128 sequential steps (kernel boundary = h dependency)
  for (int t = 0; t < TT; ++t) {
    k_lstm_step<<<8, 256, ldsEnc, stream>>>(Xenc + (size_t)t * BB * EE, EE, EE,
                                            hp[t & 1], Wenc, EE + HH, benc, Cst,
                                            hp[(t + 1) & 1]);
  }
  // T=128 even -> final encoder h lives in hp[0]; c continues in Cst

  // decoder: h history goes straight into decHs (score GEMM A operand)
  for (int t = 0; t < TT; ++t) {
    const unsigned short* hprev =
        (t == 0) ? hp[0] : (decHs + (size_t)(t - 1) * BB * HH);
    k_lstm_step<<<8, 256, ldsDec, stream>>>(Xdec + (size_t)t * BB * LDD, LDD, LDD,
                                            hprev, Wdec, LDD + HH, bdec, Cst,
                                            decHs + (size_t)t * BB * HH);
  }

  // score GEMM: 128 M-strips x 40 N-blocks, 8 waves/block, A strip in LDS
  k_score<<<128 * SC_NB, 256, (size_t)32 * HH * 2, stream>>>(decHs, Wsc,
                                                             b_score, out);
}
